// DirectVoxGO_44925357917003
// MI455X (gfx1250) — compile-verified, matrix operand order checked by reading, not run on hardware
//
#include <hip/hip_runtime.h>
#include <hip/hip_bf16.h>
#include <math.h>

typedef __attribute__((ext_vector_type(16))) _Float16 v16h;
typedef __attribute__((ext_vector_type(8)))  _Float16 v8h;
typedef __attribute__((ext_vector_type(8)))  float    v8f;

#define G        160
#define G3       (160*160*160)
#define NRAYS    4096
#define NSAMP    256
#define NPTS     (NRAYS*NSAMP)
#define K0D      12
#define K0_REC   16      // AoS record: 16 halves = 32B per voxel (12 valid + 4 zero)
#define WDIM     128
#define FEAT     72
#define FEAT_PAD 96      // K padded to multiple of 32
#define FEAT_LD  104     // LDS row stride (halves): 208B = 52 words -> conflict-free b128
#define H_LD     136     // LDS row stride (halves): 272B = 68 words -> conflict-free b128
#define TILE_PTS 128

#define ACT_SHIFT_F (-13.81550956f)  // log(1/(1-1e-6) - 1)

// ---------------- workspace layout (bytes) ----------------
#define OFF_ALPHA  ((size_t)0)                       // NPTS f32
#define OFF_WTS    (OFF_ALPHA + (size_t)NPTS*4)      // NPTS f32
#define OFF_ALAST  (OFF_WTS   + (size_t)NPTS*4)      // NRAYS f32
#define OFF_PART   (OFF_ALAST + (size_t)NRAYS*4)     // (NPTS/128)*3 f32
#define OFF_W0T    (OFF_PART  + (size_t)(NPTS/TILE_PTS)*3*4 + 256)
#define OFF_W1T    (OFF_W0T   + (size_t)WDIM*FEAT_PAD*2)
#define OFF_W2T    (OFF_W1T   + (size_t)WDIM*WDIM*2)
#define OFF_K0A    (OFF_W2T   + (size_t)16*WDIM*2)
#define WS_NEED_AOS (OFF_K0A  + (size_t)G3*K0_REC*2)

// dynamic-LDS partition (halves)
#define L_FEAT  0
#define L_H     (L_FEAT + TILE_PTS*FEAT_LD)          // 13312
#define L_W0    (L_H    + TILE_PTS*H_LD)             // +17408
#define L_W1    (L_W0   + WDIM*FEAT_LD)              // +13312
#define L_W2    (L_W1   + WDIM*H_LD)                 // +17408
#define L_END   (L_W2   + 16*H_LD)                   // +2176 = 63616 halves
#define SMEM_BYTES ((size_t)L_END * 2)               // 127232 B dynamic LDS

// 16-half fragment load from a 16B-aligned address (v16h alone would imply 32B align)
__device__ __forceinline__ v16h load16(const _Float16* p) {
    union { v16h v; v8h h[2]; } u;
    u.h[0] = *(const v8h*)p;
    u.h[1] = *(const v8h*)(p + 8);
    return u.v;
}

// ---- trilinear setup matching reference (align_corners, clamp) ----
__device__ __forceinline__ void tri_prep(float x, float y, float z,
                                         int& ix, int& iy, int& iz, float wg[8]) {
    float px = fminf(fmaxf(x, 0.f), 1.f) * 159.f;
    float py = fminf(fmaxf(y, 0.f), 1.f) * 159.f;
    float pz = fminf(fmaxf(z, 0.f), 1.f) * 159.f;
    ix = (int)fminf(floorf(px), 158.f);
    iy = (int)fminf(floorf(py), 158.f);
    iz = (int)fminf(floorf(pz), 158.f);
    float fx = px - (float)ix, fy = py - (float)iy, fz = pz - (float)iz;
    float gx = 1.f - fx, gy = 1.f - fy, gz = 1.f - fz;
    wg[0] = gx*gy*gz; wg[1] = gx*gy*fz; wg[2] = gx*fy*gz; wg[3] = gx*fy*fz;
    wg[4] = fx*gy*gz; wg[5] = fx*gy*fz; wg[6] = fx*fy*gz; wg[7] = fx*fy*fz;
}

__device__ __forceinline__ float tri_gather(const float* __restrict__ g,
                                            int base, const float wg[8]) {
    const int OY = G, OX = G * G;
    return g[base]       * wg[0] + g[base+1]       * wg[1]
         + g[base+OY]    * wg[2] + g[base+OY+1]    * wg[3]
         + g[base+OX]    * wg[4] + g[base+OX+1]    * wg[5]
         + g[base+OX+OY] * wg[6] + g[base+OX+OY+1] * wg[7];
}

// AoS gather: one (z, z+1) record pair = 64B contiguous (4x b128)
__device__ __forceinline__ void gather12_pair(const _Float16* __restrict__ k0a,
                                              int vox, float wz0, float wz1,
                                              float acc[K0D]) {
    const v8h* r = (const v8h*)(k0a + (size_t)vox * K0_REC);
    union { v8h v[4]; _Float16 h[32]; } u;
    u.v[0] = r[0]; u.v[1] = r[1]; u.v[2] = r[2]; u.v[3] = r[3];
    #pragma unroll
    for (int c = 0; c < K0D; ++c)
        acc[c] += (float)u.h[c] * wz0 + (float)u.h[K0_REC + c] * wz1;
}

// ---------------- weight pack: f32 -> f16, transposed & padded ----------------
__global__ void pack_kernel(const float* __restrict__ w0,
                            const float* __restrict__ w1,
                            const float* __restrict__ w2,
                            _Float16* __restrict__ w0T,
                            _Float16* __restrict__ w1T,
                            _Float16* __restrict__ w2T) {
    int stride = gridDim.x * blockDim.x;
    int t0 = blockIdx.x * blockDim.x + threadIdx.x;
    for (int i = t0; i < WDIM * FEAT_PAD; i += stride) {       // w0T: [WDIM][FEAT_PAD]
        int n = i / FEAT_PAD, k = i % FEAT_PAD;
        w0T[i] = (k < FEAT) ? (_Float16)w0[k * WDIM + n] : (_Float16)0.f;
    }
    for (int i = t0; i < WDIM * WDIM; i += stride) {           // w1T: [WDIM][WDIM]
        int n = i / WDIM, k = i % WDIM;
        w1T[i] = (_Float16)w1[k * WDIM + n];
    }
    for (int i = t0; i < 16 * WDIM; i += stride) {             // w2T: [16][WDIM]
        int n = i / WDIM, k = i % WDIM;
        w2T[i] = (n < 3) ? (_Float16)w2[k * 3 + n] : (_Float16)0.f;
    }
}

// ---------------- k0 SoA->AoS repack (coalesced read & write) ----------------
__global__ void k0pack_kernel(const float* __restrict__ k0,
                              _Float16* __restrict__ k0a) {
    int i = blockIdx.x * blockDim.x + threadIdx.x;   // voxel id
    if (i >= G3) return;
    _Float16 rec[K0_REC];
    #pragma unroll
    for (int c = 0; c < K0D; ++c) rec[c] = (_Float16)k0[(size_t)c * G3 + i];
    #pragma unroll
    for (int c = K0D; c < K0_REC; ++c) rec[c] = (_Float16)0.f;
    v8h* out = (v8h*)(k0a + (size_t)i * K0_REC);
    out[0] = *(v8h*)rec;
    out[1] = *(v8h*)(rec + 8);
}

// ---------------- density -> alpha per point ----------------
__global__ void alpha_kernel(const float* __restrict__ pts,
                             const float* __restrict__ dens,
                             float* __restrict__ alpha) {
    int i = blockIdx.x * blockDim.x + threadIdx.x;
    if (i >= NPTS) return;
    const float* p = pts + (size_t)i * 3;
    int ix, iy, iz; float wg[8];
    tri_prep(p[0], p[1], p[2], ix, iy, iz, wg);
    float d = tri_gather(dens, (ix * G + iy) * G + iz, wg);
    float xx = d + ACT_SHIFT_F;
    float sp = fmaxf(xx, 0.f) + log1pf(__expf(-fabsf(xx)));   // softplus
    alpha[i] = 1.f - __expf(-sp);
}

// ---------------- per-ray cumprod scan ----------------
__global__ void scan_kernel(const float* __restrict__ alpha,
                            float* __restrict__ wts,
                            float* __restrict__ alast) {
    int r = blockIdx.x * blockDim.x + threadIdx.x;
    if (r >= NRAYS) return;
    const float* a = alpha + (size_t)r * NSAMP;
    float* w = wts + (size_t)r * NSAMP;
    float cum = 1.f;
    #pragma unroll 4
    for (int s = 0; s < NSAMP; ++s) {
        float al = a[s];
        w[s] = al * cum;
        cum *= fmaxf(1.f - al, 1e-10f);
    }
    alast[r] = cum;
}

// ---------------- fused feature + WMMA MLP + weighted reduce ----------------
// Occupancy is LDS-bound (127 KB -> 2 blocks/WGP); tell the compiler so it
// spends VGPRs on keeping all 8 B fragments live per k-step.
template <bool AOS>
__global__ __launch_bounds__(256, 2)
void DirectVoxGO_mlp_kernel(const float* __restrict__ pts,
                            const float* __restrict__ vdirs,
                            const float* __restrict__ k0,
                            const _Float16* __restrict__ k0a,
                            const _Float16* __restrict__ w0T,
                            const _Float16* __restrict__ w1T,
                            const _Float16* __restrict__ w2T,
                            const float* __restrict__ b0,
                            const float* __restrict__ b1,
                            const float* __restrict__ b2,
                            const float* __restrict__ wts,
                            float* __restrict__ partials) {
    extern __shared__ _Float16 smem[];
    _Float16* featLDS = smem + L_FEAT;
    _Float16* hLDS    = smem + L_H;
    _Float16* w0L     = smem + L_W0;
    _Float16* w1L     = smem + L_W1;
    _Float16* w2L     = smem + L_W2;
    __shared__ float wtLDS[TILE_PTS];
    __shared__ float red[256];

    const int tid = threadIdx.x;
    const int ptBase = blockIdx.x * TILE_PTS;
    const int lp = tid & 127;
    const int gp = ptBase + lp;

    // ---------- stage weights to LDS (b128 copies; padded conflict-free strides) ----------
    {
        const v8h* s0 = (const v8h*)w0T;  v8h* d0 = (v8h*)w0L;   // 12 -> 13 v8h/row
        for (int i = tid; i < WDIM * 12; i += 256) {
            int n = i / 12, k = i - n * 12;
            d0[n * 13 + k] = s0[i];
        }
        const v8h* s1 = (const v8h*)w1T;  v8h* d1 = (v8h*)w1L;   // 16 -> 17 v8h/row
        for (int i = tid; i < WDIM * 16; i += 256) {
            int n = i >> 4, k = i & 15;
            d1[n * 17 + k] = s1[i];
        }
        const v8h* s2 = (const v8h*)w2T;  v8h* d2 = (v8h*)w2L;
        for (int i = tid; i < 16 * 16; i += 256) {
            int n = i >> 4, k = i & 15;
            d2[n * 17 + k] = s2[i];
        }
    }

    // ---------- feature phase ----------
    if (tid < 128) {
        const float* p = pts + (size_t)gp * 3;
        float x = p[0], y = p[1], z = p[2];
        int ix, iy, iz; float wg[8];
        tri_prep(x, y, z, ix, iy, iz, wg);
        _Float16* f = &featLDS[lp * FEAT_LD];
        if constexpr (AOS) {
            float acc[K0D];
            #pragma unroll
            for (int c = 0; c < K0D; ++c) acc[c] = 0.f;
            int b00 = (ix * G + iy) * G + iz;
            gather12_pair(k0a, b00,             wg[0], wg[1], acc);  // (x0,y0)
            gather12_pair(k0a, b00 + G,         wg[2], wg[3], acc);  // (x0,y1)
            gather12_pair(k0a, b00 + G * G,     wg[4], wg[5], acc);  // (x1,y0)
            gather12_pair(k0a, b00 + G * G + G, wg[6], wg[7], acc);  // (x1,y1)
            #pragma unroll
            for (int c = 0; c < K0D; ++c) f[c] = (_Float16)acc[c];
        } else {
            int base = (ix * G + iy) * G + iz;
            #pragma unroll
            for (int c = 0; c < K0D; ++c)
                f[c] = (_Float16)tri_gather(k0 + (size_t)c * G3, base, wg);
        }
        f[12] = (_Float16)x; f[13] = (_Float16)y; f[14] = (_Float16)z;
        #pragma unroll
        for (int j = 0; j < 5; ++j) {
            float fr = (float)(1 << j);
            f[15 + j] = (_Float16)__sinf(x * fr);
            f[20 + j] = (_Float16)__sinf(y * fr);
            f[25 + j] = (_Float16)__sinf(z * fr);
            f[30 + j] = (_Float16)__cosf(x * fr);
            f[35 + j] = (_Float16)__cosf(y * fr);
            f[40 + j] = (_Float16)__cosf(z * fr);
        }
    } else {
        int ray = gp >> 8;
        const float* v = vdirs + (size_t)ray * 3;
        float vx = v[0], vy = v[1], vz = v[2];
        _Float16* f = &featLDS[lp * FEAT_LD];
        f[45] = (_Float16)vx; f[46] = (_Float16)vy; f[47] = (_Float16)vz;
        #pragma unroll
        for (int j = 0; j < 4; ++j) {
            float fr = (float)(1 << j);
            f[48 + j] = (_Float16)__sinf(vx * fr);
            f[52 + j] = (_Float16)__sinf(vy * fr);
            f[56 + j] = (_Float16)__sinf(vz * fr);
            f[60 + j] = (_Float16)__cosf(vx * fr);
            f[64 + j] = (_Float16)__cosf(vy * fr);
            f[68 + j] = (_Float16)__cosf(vz * fr);
        }
        #pragma unroll
        for (int i = FEAT; i < FEAT_LD; ++i) f[i] = (_Float16)0.f;
        wtLDS[lp] = wts[gp];
    }
    __syncthreads();

    // ---------- WMMA phase: each wave owns a 16-point row tile ----------
    const int lane = tid & 31, wave = tid >> 5;
    const int hi = lane >> 4, ln = lane & 15;
    const int rowBase = wave * 16;

    // ---- layer 0: [16 x 96] @ [96 x 128]; batch all 8 B fragments per k-step ----
    v8f acc[8];
    #pragma unroll
    for (int n = 0; n < 8; ++n)
        #pragma unroll
        for (int i = 0; i < 8; ++i) acc[n][i] = 0.f;

    #pragma unroll
    for (int kt = 0; kt < 3; ++kt) {
        v16h a = load16(&featLDS[(rowBase + ln) * FEAT_LD + kt * 32 + hi * 16]);
        v16h bf[8];
        #pragma unroll
        for (int n = 0; n < 8; ++n)
            bf[n] = load16(&w0L[(n * 16 + ln) * FEAT_LD + kt * 32 + hi * 16]);
        #pragma unroll
        for (int n = 0; n < 8; ++n)
            acc[n] = __builtin_amdgcn_wmma_f32_16x16x32_f16(
                false, a, false, bf[n], (short)0, acc[n], false, false);
    }
    #pragma unroll
    for (int n = 0; n < 8; ++n) {
        int col = n * 16 + ln;
        float bias = b0[col];
        #pragma unroll
        for (int v = 0; v < 8; ++v) {
            float h = fmaxf(acc[n][v] + bias, 0.f);
            hLDS[(rowBase + v + hi * 8) * H_LD + col] = (_Float16)h;
        }
    }
    __syncthreads();

    // ---- layer 1: [16 x 128] @ [128 x 128]; batched B fragments ----
    v8f acc2[8];
    #pragma unroll
    for (int n = 0; n < 8; ++n)
        #pragma unroll
        for (int i = 0; i < 8; ++i) acc2[n][i] = 0.f;

    #pragma unroll
    for (int kt = 0; kt < 4; ++kt) {
        v16h a = load16(&hLDS[(rowBase + ln) * H_LD + kt * 32 + hi * 16]);
        v16h bf[8];
        #pragma unroll
        for (int n = 0; n < 8; ++n)
            bf[n] = load16(&w1L[(n * 16 + ln) * H_LD + kt * 32 + hi * 16]);
        #pragma unroll
        for (int n = 0; n < 8; ++n)
            acc2[n] = __builtin_amdgcn_wmma_f32_16x16x32_f16(
                false, a, false, bf[n], (short)0, acc2[n], false, false);
    }
    __syncthreads();  // WAR: layer-1 reads complete before overwriting hLDS
    #pragma unroll
    for (int n = 0; n < 8; ++n) {
        int col = n * 16 + ln;
        float bias = b1[col];
        #pragma unroll
        for (int v = 0; v < 8; ++v) {
            float h = fmaxf(acc2[n][v] + bias, 0.f);
            hLDS[(rowBase + v + hi * 8) * H_LD + col] = (_Float16)h;
        }
    }
    __syncthreads();

    // ---- layer 2: [16 x 128] @ [128 x 16]; batch all 4 A + 4 B fragments ----
    v8f acc3;
    #pragma unroll
    for (int i = 0; i < 8; ++i) acc3[i] = 0.f;
    {
        v16h a2[4], bf2[4];
        #pragma unroll
        for (int kt = 0; kt < 4; ++kt) {
            a2[kt]  = load16(&hLDS[(rowBase + ln) * H_LD + kt * 32 + hi * 16]);
            bf2[kt] = load16(&w2L[ln * H_LD + kt * 32 + hi * 16]);
        }
        #pragma unroll
        for (int kt = 0; kt < 4; ++kt)
            acc3 = __builtin_amdgcn_wmma_f32_16x16x32_f16(
                false, a2[kt], false, bf2[kt], (short)0, acc3, false, false);
    }
    float bias2 = (ln < 3) ? b2[ln] : 0.f;
    float s = 0.f;
    #pragma unroll
    for (int v = 0; v < 8; ++v) {
        float xx = acc3[v] + bias2;
        float sg = 1.f / (1.f + __expf(-xx));       // sigmoid
        s += sg * wtLDS[rowBase + v + hi * 8];      // ray-march weight
    }
    red[tid] = s;
    __syncthreads();

    if (tid < 3) {  // deterministic in-block reduction
        float t = 0.f;
        #pragma unroll
        for (int wv = 0; wv < 8; ++wv)
            t += red[wv * 32 + tid] + red[wv * 32 + 16 + tid];
        partials[blockIdx.x * 3 + tid] = t;
    }
}

// ---------------- final per-ray combine ----------------
__global__ void finish_kernel(const float* __restrict__ alast,
                              const float* __restrict__ part,
                              float* __restrict__ out) {
    int r = blockIdx.x * blockDim.x + threadIdx.x;
    if (r >= NRAYS) return;
    float last = alast[r];
    #pragma unroll
    for (int c = 0; c < 3; ++c)
        out[r * 3 + c] = last + part[(2 * r) * 3 + c] + part[(2 * r + 1) * 3 + c];
}

extern "C" void kernel_launch(void* const* d_in, const int* in_sizes, int n_in,
                              void* d_out, int out_size, void* d_ws, size_t ws_size,
                              hipStream_t stream) {
    const float* rays_pts = (const float*)d_in[0];
    const float* viewdirs = (const float*)d_in[1];
    const float* density  = (const float*)d_in[2];
    const float* k0       = (const float*)d_in[3];
    const float* w0       = (const float*)d_in[4];
    const float* b0       = (const float*)d_in[5];
    const float* w1       = (const float*)d_in[6];
    const float* b1       = (const float*)d_in[7];
    const float* w2       = (const float*)d_in[8];
    const float* b2       = (const float*)d_in[9];

    char* ws = (char*)d_ws;
    float*    alpha    = (float*)(ws + OFF_ALPHA);
    float*    weights  = (float*)(ws + OFF_WTS);
    float*    alast    = (float*)(ws + OFF_ALAST);
    float*    partials = (float*)(ws + OFF_PART);
    _Float16* w0T      = (_Float16*)(ws + OFF_W0T);
    _Float16* w1T      = (_Float16*)(ws + OFF_W1T);
    _Float16* w2T      = (_Float16*)(ws + OFF_W2T);
    _Float16* k0a      = (_Float16*)(ws + OFF_K0A);

    pack_kernel<<<32, 256, 0, stream>>>(w0, w1, w2, w0T, w1T, w2T);
    alpha_kernel<<<NPTS / 256, 256, 0, stream>>>(rays_pts, density, alpha);
    scan_kernel<<<NRAYS / 256, 256, 0, stream>>>(alpha, weights, alast);

    const bool aos = (ws_size >= WS_NEED_AOS);   // deterministic per ws_size
    if (aos) {
        k0pack_kernel<<<G3 / 256, 256, 0, stream>>>(k0, k0a);
        DirectVoxGO_mlp_kernel<true><<<NPTS / TILE_PTS, 256, SMEM_BYTES, stream>>>(
            rays_pts, viewdirs, k0, k0a, w0T, w1T, w2T, b0, b1, b2, weights, partials);
    } else {
        DirectVoxGO_mlp_kernel<false><<<NPTS / TILE_PTS, 256, SMEM_BYTES, stream>>>(
            rays_pts, viewdirs, k0, k0a, w0T, w1T, w2T, b0, b1, b2, weights, partials);
    }
    finish_kernel<<<NRAYS / 256, 256, 0, stream>>>(alast, partials, (float*)d_out);
}
